// CellTypeSpecEmbedding_8375186227866
// MI455X (gfx1250) — matrix operand order repaired and checked by reading, not compiled, not run on hardware
//
#include <hip/hip_runtime.h>
#include <math.h>

typedef __attribute__((ext_vector_type(16))) __bf16 v16bf;
typedef __attribute__((ext_vector_type(8)))  float  v8f;
typedef unsigned short bf16_t;

union FragU { unsigned u[8]; v16bf v; };

__device__ __forceinline__ bf16_t f2bf(float f) {
    union { float f; unsigned u; } c; c.f = f;
    unsigned u = c.u;
    unsigned r = (u + 0x7FFFu + ((u >> 16) & 1u)) >> 16;
    return (bf16_t)r;
}
__device__ __forceinline__ unsigned pk2(float a, float b) {
    return (unsigned)f2bf(a) | ((unsigned)f2bf(b) << 16);
}

// ---- WMMA fragment loaders (16x16x32 bf16; this code's contiguous k order) ----
// A 16x32: lane m=lane&15, half=lane>>4; dword d -> K = 16*half + 2*d
__device__ __forceinline__ v16bf ldsA_f32(const float* base, int row, int stride, int kb, int hf) {
    FragU f; const float* rp = base + row * stride + kb + 16 * hf;
#pragma unroll
    for (int d = 0; d < 8; ++d) {
        int k0 = 2 * d;
        f.u[d] = pk2(rp[k0], rp[k0 + 1]);
    }
    return f.v;
}
__device__ __forceinline__ v16bf ldsA_b16(const bf16_t* base, int row, int stride, int kb, int hf) {
    FragU f; const unsigned* p32 = (const unsigned*)base;
    int off = row * stride + kb + 16 * hf;  // even
#pragma unroll
    for (int d = 0; d < 8; ++d) {
        int k0 = 2 * d;
        f.u[d] = p32[(off + k0) >> 1];
    }
    return f.v;
}
// B 32x16: lane n=lane&15, half=lane>>4; dword d -> K = 16*half + 2*d
__device__ __forceinline__ v16bf ldsB_f32(const float* base, int col, int stride, int kb, int hf) {
    FragU f; const float* rp = base + col * stride + kb + 16 * hf;
#pragma unroll
    for (int d = 0; d < 8; ++d) f.u[d] = pk2(rp[2 * d], rp[2 * d + 1]);
    return f.v;
}
__device__ __forceinline__ v16bf glbB_b16(const bf16_t* base, int col, int Kdim, int kb, int hf) {
    FragU f; const unsigned* p32 = (const unsigned*)base;
    int off = (col * Kdim + kb + 16 * hf) >> 1;
#pragma unroll
    for (int d = 0; d < 8; ++d) f.u[d] = p32[off + d];
    return f.v;
}
__device__ __forceinline__ v8f wmma_bf16(v16bf a, v16bf b, v8f c) {
    return __builtin_amdgcn_wmma_f32_16x16x32_bf16(false, a, false, b, (short)0, c, false, false);
}

// ---- constants ----
#define BATCH 64
#define GG    4
#define KK    128
#define DD    256
#define HH    8
#define DH    32
#define LL    3
#define DFF   1024
#define XFS   257            // padded f32 stride for activations
// LDS float offsets
#define OFF_XF   0           // [128][257]            32896
#define OFF_CTX  32896       // bf16 [128][258] / HID bf16 [32][1028]   (16512 floats)
#define OFF_Q    49408       // [128][33]  4224
#define OFF_KB   53632       // [128][33]  4224
#define OFF_VT   57856       // [32][129]  4128
#define OFF_LG   61984       // [128][129] 16512
#define OFF_SPW1 78496
#define OFF_SPB1 78560
#define OFF_SPW2 78624
#define OFF_SPB2 79136
#define LDS_FLOATS 79144
#define LDS_BYTES (LDS_FLOATS * 4)

struct GParams {
    const int   *node_id, *neigh, *deg;
    const float *feat, *centr;
    const float *lin_b, *bq, *bk, *bv, *bo;
    const float *spW1, *spb1, *spW2, *spb2;
    const float *ln1g, *ln1b, *fb1, *fb2, *ln2g, *ln2b;
    const float *spatial;
    const bf16_t *linWt, *Wqt, *Wkt, *Wvt, *Wot, *fW1t, *fW2t;
    float *embs;
};

__device__ __forceinline__ void layer_norm_rows(float* XF, const float* gg, const float* bb, int tid) {
    if (tid < KK) {
        float* row = XF + tid * XFS;
        float s = 0.f;
        for (int d = 0; d < DD; ++d) s += row[d];
        float m = s * (1.0f / DD);
        float v = 0.f;
        for (int d = 0; d < DD; ++d) { float t = row[d] - m; v += t * t; }
        float inv = rsqrtf(v * (1.0f / DD) + 1e-5f);
        for (int d = 0; d < DD; ++d) row[d] = (row[d] - m) * inv * gg[d] + bb[d];
    }
}

__global__ __launch_bounds__(256) void graphormer_mega(GParams p) {
    extern __shared__ float lds[];
    float*  XF  = lds + OFF_XF;
    bf16_t* CTX = (bf16_t*)(lds + OFF_CTX);
    float*  Q   = lds + OFF_Q;
    float*  KB  = lds + OFF_KB;
    float*  VT  = lds + OFF_VT;
    float*  LG  = lds + OFF_LG;
    float*  SPW1 = lds + OFF_SPW1;
    float*  SPB1 = lds + OFF_SPB1;
    float*  SPW2 = lds + OFF_SPW2;
    float*  SPB2 = lds + OFF_SPB2;

    const int tid = threadIdx.x, lane = tid & 31, wave = tid >> 5;
    const int l15 = lane & 15, hf = (lane >> 4) & 1;
    const int bg = blockIdx.x, b = bg >> 2, g = bg & 3;
    const int node = p.node_id[b];
    const float* hopbase = p.spatial + (size_t)(node * GG + g) * KK * KK;

    // ---- prologue: XIN(bf16, in CTX) = nan_to_num(feat)*sqrt(D) + centr[deg] ----
    for (int idx = tid; idx < KK * DD; idx += 256) {
        int r = idx >> 8, d = idx & 255;
        int nidx = p.neigh[node * (GG * KK) + g * KK + r];
        float f = p.feat[(size_t)nidx * DD + d];
        if (!(f == f)) f = 0.f;
        int dg = p.deg[nidx];
        float c = p.centr[(size_t)dg * DD + d];
        CTX[r * 258 + d] = f2bf(f * 16.0f + c);
    }
    __syncthreads();

    // ---- XF = relu(XIN @ linW + lin_b) : M=128,N=256,K=256 ----
    for (int t = wave; t < 128; t += 8) {
        int mb = (t & 7) * 16, nb = (t >> 3) * 16;
        v8f acc = {0.f, 0.f, 0.f, 0.f, 0.f, 0.f, 0.f, 0.f};
        for (int kb = 0; kb < DD; kb += 32)
            acc = wmma_bf16(ldsA_b16(CTX, mb + l15, 258, kb, hf),
                            glbB_b16(p.linWt, nb + l15, DD, kb, hf), acc);
        int n = nb + l15; float bias = p.lin_b[n];
#pragma unroll
        for (int v = 0; v < 8; ++v) {
            float r = acc[v] + bias;
            XF[(mb + v + 8 * hf) * XFS + n] = r > 0.f ? r : 0.f;
        }
    }
    __syncthreads();

    for (int l = 0; l < LL; ++l) {
        // spatial-MLP tables to LDS
        if (tid < 64) { SPW1[tid] = p.spW1[l * 64 + tid]; SPB1[tid] = p.spb1[l * 64 + tid]; }
        for (int i = tid; i < 512; i += 256) SPW2[i] = p.spW2[l * 512 + i];
        if (tid < 8) SPB2[tid] = p.spb2[l * 8 + tid];
        __syncthreads();

        const bf16_t* Wqt = p.Wqt + l * 65536;
        const bf16_t* Wkt = p.Wkt + l * 65536;
        const bf16_t* Wvt = p.Wvt + l * 65536;
        const bf16_t* Wot = p.Wot + l * 65536;
        const float *bq = p.bq + l * DD, *bk = p.bk + l * DD, *bv = p.bv + l * DD, *bo = p.bo + l * DD;

        for (int h = 0; h < HH; ++h) {
            // Q/K: [128][33] f32 ; V stored transposed VT[32][129]
            for (int t = wave; t < 16; t += 8) {
                int mb = (t & 7) * 16, nb = (t >> 3) * 16;
                v8f acc = {0.f, 0.f, 0.f, 0.f, 0.f, 0.f, 0.f, 0.f};
                for (int kb = 0; kb < DD; kb += 32)
                    acc = wmma_bf16(ldsA_f32(XF, mb + l15, XFS, kb, hf),
                                    glbB_b16(Wqt, h * DH + nb + l15, DD, kb, hf), acc);
                int n = nb + l15; float bias = bq[h * DH + n];
#pragma unroll
                for (int v = 0; v < 8; ++v) Q[(mb + v + 8 * hf) * 33 + n] = acc[v] + bias;
            }
            for (int t = wave; t < 16; t += 8) {
                int mb = (t & 7) * 16, nb = (t >> 3) * 16;
                v8f acc = {0.f, 0.f, 0.f, 0.f, 0.f, 0.f, 0.f, 0.f};
                for (int kb = 0; kb < DD; kb += 32)
                    acc = wmma_bf16(ldsA_f32(XF, mb + l15, XFS, kb, hf),
                                    glbB_b16(Wkt, h * DH + nb + l15, DD, kb, hf), acc);
                int n = nb + l15; float bias = bk[h * DH + n];
#pragma unroll
                for (int v = 0; v < 8; ++v) KB[(mb + v + 8 * hf) * 33 + n] = acc[v] + bias;
            }
            for (int t = wave; t < 16; t += 8) {
                int mb = (t & 7) * 16, nb = (t >> 3) * 16;
                v8f acc = {0.f, 0.f, 0.f, 0.f, 0.f, 0.f, 0.f, 0.f};
                for (int kb = 0; kb < DD; kb += 32)
                    acc = wmma_bf16(ldsA_f32(XF, mb + l15, XFS, kb, hf),
                                    glbB_b16(Wvt, h * DH + nb + l15, DD, kb, hf), acc);
                int n = nb + l15; float bias = bv[h * DH + n];
#pragma unroll
                for (int v = 0; v < 8; ++v) VT[n * 129 + (mb + v + 8 * hf)] = acc[v] + bias;
            }
            __syncthreads();

            // logits = Q@K^T/sqrt(DH) + spatialMLP(hop) + mask : M=128,N=128,K=32
            for (int t = wave; t < 64; t += 8) {
                int mb = (t & 7) * 16, nb = (t >> 3) * 16;
                v8f acc = {0.f, 0.f, 0.f, 0.f, 0.f, 0.f, 0.f, 0.f};
                acc = wmma_bf16(ldsA_f32(Q, mb + l15, 33, 0, hf),
                                ldsB_f32(KB, nb + l15, 33, 0, hf), acc);
                int n = nb + l15;
#pragma unroll
                for (int v = 0; v < 8; ++v) {
                    int m = mb + v + 8 * hf;
                    float hop = hopbase[m * KK + n];
                    float s = SPB2[h];
                    for (int j = 0; j < 64; ++j) {
                        float rr = hop * SPW1[j] + SPB1[j];
                        if (rr > 0.f) s += rr * SPW2[j * 8 + h];
                    }
                    float lg = acc[v] * 0.17677669529663687f + s;
                    if (hop == -1.0f) lg -= 1e9f;
                    LG[m * 129 + n] = lg;
                }
            }
            __syncthreads();

            // softmax rows (row-per-thread, odd stride => conflict-free)
            if (tid < KK) {
                float* row = LG + tid * 129;
                float mx = row[0];
                for (int j = 1; j < KK; ++j) mx = fmaxf(mx, row[j]);
                float sum = 0.f;
                for (int j = 0; j < KK; ++j) { float e = __expf(row[j] - mx); row[j] = e; sum += e; }
                float inv = 1.0f / sum;
                for (int j = 0; j < KK; ++j) row[j] *= inv;
            }
            __syncthreads();

            // ctx_h = attn @ v : M=128,N=32,K=128 -> CTX[:, h*32:+32] bf16
            for (int t = wave; t < 16; t += 8) {
                int mb = (t & 7) * 16, nb = (t >> 3) * 16;
                v8f acc = {0.f, 0.f, 0.f, 0.f, 0.f, 0.f, 0.f, 0.f};
                for (int kb = 0; kb < KK; kb += 32)
                    acc = wmma_bf16(ldsA_f32(LG, mb + l15, 129, kb, hf),
                                    ldsB_f32(VT, nb + l15, 129, kb, hf), acc);
                int n = nb + l15;
#pragma unroll
                for (int v = 0; v < 8; ++v)
                    CTX[(mb + v + 8 * hf) * 258 + h * DH + n] = f2bf(acc[v]);
            }
            __syncthreads();
        }

        // XF += CTX @ Wo + bo ; then LN1
        for (int t = wave; t < 128; t += 8) {
            int mb = (t & 7) * 16, nb = (t >> 3) * 16;
            v8f acc = {0.f, 0.f, 0.f, 0.f, 0.f, 0.f, 0.f, 0.f};
            for (int kb = 0; kb < DD; kb += 32)
                acc = wmma_bf16(ldsA_b16(CTX, mb + l15, 258, kb, hf),
                                glbB_b16(Wot, nb + l15, DD, kb, hf), acc);
            int n = nb + l15; float bias = bo[n];
#pragma unroll
            for (int v = 0; v < 8; ++v) XF[(mb + v + 8 * hf) * XFS + n] += acc[v] + bias;
        }
        __syncthreads();
        layer_norm_rows(XF, p.ln1g + l * DD, p.ln1b + l * DD, tid);
        __syncthreads();

        // FFN, M-chunked by 32 rows; hidden chunk bf16 [32][1028] in CTX region
        const bf16_t* fW1t = p.fW1t + l * (DD * DFF);
        const bf16_t* fW2t = p.fW2t + l * (DD * DFF);
        const float *fb1 = p.fb1 + l * DFF, *fb2 = p.fb2 + l * DD;
        for (int c = 0; c < 4; ++c) {
            int rb = c * 32;
            // HID = relu(XF_chunk @ fW1 + fb1): M=32,N=1024,K=256
            for (int t = wave; t < 128; t += 8) {
                int tm = t & 1, nb = (t >> 1) * 16;
                v8f acc = {0.f, 0.f, 0.f, 0.f, 0.f, 0.f, 0.f, 0.f};
                for (int kb = 0; kb < DD; kb += 32)
                    acc = wmma_bf16(ldsA_f32(XF, rb + tm * 16 + l15, XFS, kb, hf),
                                    glbB_b16(fW1t, nb + l15, DD, kb, hf), acc);
                int n = nb + l15; float bias = fb1[n];
#pragma unroll
                for (int v = 0; v < 8; ++v) {
                    float r = acc[v] + bias;
                    CTX[(tm * 16 + v + 8 * hf) * 1028 + n] = f2bf(r > 0.f ? r : 0.f);
                }
            }
            __syncthreads();
            // XF_chunk += HID @ fW2 + fb2 : M=32,N=256,K=1024
            for (int t = wave; t < 32; t += 8) {
                int tm = t & 1, nb = (t >> 1) * 16;
                v8f acc = {0.f, 0.f, 0.f, 0.f, 0.f, 0.f, 0.f, 0.f};
                for (int kb = 0; kb < DFF; kb += 32)
                    acc = wmma_bf16(ldsA_b16(CTX, tm * 16 + l15, 1028, kb, hf),
                                    glbB_b16(fW2t, nb + l15, DFF, kb, hf), acc);
                int n = nb + l15; float bias = fb2[n];
#pragma unroll
                for (int v = 0; v < 8; ++v)
                    XF[(rb + tm * 16 + v + 8 * hf) * XFS + n] += acc[v] + bias;
            }
            __syncthreads();
        }
        layer_norm_rows(XF, p.ln2g + l * DD, p.ln2b + l * DD, tid);
        __syncthreads();
    }

    // emit row 0
    p.embs[bg * DD + tid] = XF[tid];
}

// ---- transpose + bf16 convert: src (K,N) row-major -> dst (N,K) row-major bf16 ----
__global__ void conv_t_bf16(const float* src, bf16_t* dst, int K, int N) {
    int i = blockIdx.x * blockDim.x + threadIdx.x;
    if (i < K * N) {
        int n = i / K, k = i - n * K;
        dst[i] = f2bf(src[(size_t)k * N + n]);
    }
}

// ---- fusion: scores = tanh(x@fusW)@fusv ; softmax over G ; weighted sum ----
__global__ __launch_bounds__(256) void fuse_kernel(const float* embs, const float* fusW,
                                                   const float* fusv, float* out) {
    __shared__ float xs[GG][DD];
    __shared__ float ts[GG][DD];
    __shared__ float red[256];
    int b = blockIdx.x, t = threadIdx.x;
    for (int g = 0; g < GG; ++g) xs[g][t] = embs[(b * GG + g) * DD + t];
    __syncthreads();
    for (int g = 0; g < GG; ++g) {
        float a = 0.f;
        for (int k = 0; k < DD; ++k) a += xs[g][k] * fusW[k * DD + t];
        ts[g][t] = tanhf(a);
    }
    __syncthreads();
    float sc[GG];
    for (int g = 0; g < GG; ++g) {
        red[t] = ts[g][t] * fusv[t];
        __syncthreads();
        for (int s = 128; s > 0; s >>= 1) { if (t < s) red[t] += red[t + s]; __syncthreads(); }
        sc[g] = red[0];
        __syncthreads();
    }
    float mx = sc[0];
    for (int g = 1; g < GG; ++g) mx = fmaxf(mx, sc[g]);
    float sum = 0.f, w[GG];
    for (int g = 0; g < GG; ++g) { w[g] = __expf(sc[g] - mx); sum += w[g]; }
    float inv = 1.0f / sum, o = 0.f;
    for (int g = 0; g < GG; ++g) o += (w[g] * inv) * xs[g][t];
    out[b * DD + t] = o;
}

extern "C" void kernel_launch(void* const* d_in, const int* in_sizes, int n_in,
                              void* d_out, int out_size, void* d_ws, size_t ws_size,
                              hipStream_t stream) {
    (void)in_sizes; (void)n_in; (void)out_size; (void)ws_size;
    const int*   node_id = (const int*)d_in[0];
    const int*   neigh   = (const int*)d_in[1];
    const int*   deg     = (const int*)d_in[2];
    const float* feat    = (const float*)d_in[3];
    const float* centr   = (const float*)d_in[4];
    const float* lin_W   = (const float*)d_in[5];
    const float* lin_b   = (const float*)d_in[6];
    const float* Wq      = (const float*)d_in[7];
    const float* bq      = (const float*)d_in[8];
    const float* Wk      = (const float*)d_in[9];
    const float* bk      = (const float*)d_in[10];
    const float* Wv      = (const float*)d_in[11];
    const float* bv      = (const float*)d_in[12];
    const float* Wo      = (const float*)d_in[13];
    const float* bo      = (const float*)d_in[14];
    const float* spW1    = (const float*)d_in[15];
    const float* spb1    = (const float*)d_in[16];
    const float* spW2    = (const float*)d_in[17];
    const float* spb2    = (const float*)d_in[18];
    const float* ln1g    = (const float*)d_in[19];
    const float* ln1b    = (const float*)d_in[20];
    const float* fW1     = (const float*)d_in[21];
    const float* fb1     = (const float*)d_in[22];
    const float* fW2     = (const float*)d_in[23];
    const float* fb2     = (const float*)d_in[24];
    const float* ln2g    = (const float*)d_in[25];
    const float* ln2b    = (const float*)d_in[26];
    const float* fusW    = (const float*)d_in[27];
    const float* fusv    = (const float*)d_in[28];
    const float* spatial = (const float*)d_in[29];

    char* ws = (char*)d_ws;
    size_t off = 0;
    float* embs = (float*)(ws + off); off += (size_t)BATCH * GG * DD * 4;
    bf16_t* linWt = (bf16_t*)(ws + off); off += (size_t)65536 * 2;
    bf16_t* Wqt = (bf16_t*)(ws + off); off += (size_t)3 * 65536 * 2;
    bf16_t* Wkt = (bf16_t*)(ws + off); off += (size_t)3 * 65536 * 2;
    bf16_t* Wvt = (bf16_t*)(ws + off); off += (size_t)3 * 65536 * 2;
    bf16_t* Wot = (bf16_t*)(ws + off); off += (size_t)3 * 65536 * 2;
    bf16_t* fW1t = (bf16_t*)(ws + off); off += (size_t)3 * 262144 * 2;
    bf16_t* fW2t = (bf16_t*)(ws + off); off += (size_t)3 * 262144 * 2;

    conv_t_bf16<<<256, 256, 0, stream>>>(lin_W, linWt, 256, 256);
    for (int l = 0; l < 3; ++l) {
        conv_t_bf16<<<256, 256, 0, stream>>>(Wq + l * 65536, Wqt + l * 65536, 256, 256);
        conv_t_bf16<<<256, 256, 0, stream>>>(Wk + l * 65536, Wkt + l * 65536, 256, 256);
        conv_t_bf16<<<256, 256, 0, stream>>>(Wv + l * 65536, Wvt + l * 65536, 256, 256);
        conv_t_bf16<<<256, 256, 0, stream>>>(Wo + l * 65536, Wot + l * 65536, 256, 256);
        conv_t_bf16<<<1024, 256, 0, stream>>>(fW1 + l * 262144, fW1t + l * 262144, 256, 1024);
        conv_t_bf16<<<1024, 256, 0, stream>>>(fW2 + l * 262144, fW2t + l * 262144, 1024, 256);
    }

    GParams P;
    P.node_id = node_id; P.neigh = neigh; P.deg = deg;
    P.feat = feat; P.centr = centr;
    P.lin_b = lin_b; P.bq = bq; P.bk = bk; P.bv = bv; P.bo = bo;
    P.spW1 = spW1; P.spb1 = spb1; P.spW2 = spW2; P.spb2 = spb2;
    P.ln1g = ln1g; P.ln1b = ln1b; P.fb1 = fb1; P.fb2 = fb2; P.ln2g = ln2g; P.ln2b = ln2b;
    P.spatial = spatial;
    P.linWt = linWt; P.Wqt = Wqt; P.Wkt = Wkt; P.Wvt = Wvt; P.Wot = Wot;
    P.fW1t = fW1t; P.fW2t = fW2t;
    P.embs = embs;

    hipFuncSetAttribute((const void*)graphormer_mega,
                        hipFuncAttributeMaxDynamicSharedMemorySize, LDS_BYTES);
    graphormer_mega<<<BATCH * GG, 256, LDS_BYTES, stream>>>(P);
    fuse_kernel<<<BATCH, 256, 0, stream>>>(embs, fusW, fusv, (float*)d_out);
}
